// AttentionNet_40364102647822
// MI455X (gfx1250) — matrix operand order repaired and checked by reading, not compiled, not run on hardware
//
#include <hip/hip_runtime.h>

// ---- shapes ----
#define ENC_DIMM 2048
#define ATT_DIMM 512
#define BATCH    256
#define PIX      196
#define MROWS    (BATCH * PIX)   // 50176 (divisible by 64: 784 tiles)
#define MTILE    64
#define KC       128
#define LDS_STR  (KC + 8)        // 136 bf16 -> 272B row stride (16B multiple, bank-staggered)

typedef __attribute__((ext_vector_type(16))) __bf16 bf16x16;
typedef __attribute__((ext_vector_type(8)))  __bf16 bf16x8;
typedef __attribute__((ext_vector_type(4)))  __bf16 bf16x4;
typedef __attribute__((ext_vector_type(8)))  float  f32x8;
typedef unsigned int v4u __attribute__((ext_vector_type(4)));
typedef int          v8i __attribute__((ext_vector_type(8)));
typedef int          v4i __attribute__((ext_vector_type(4)));

// ---------------------------------------------------------------------------
// Kernel 1: one-time W_enc f32 -> bf16 (2 MB copy, stays hot in 192MB L2)
// ---------------------------------------------------------------------------
__global__ __launch_bounds__(256) void k_wenc_bf16(const float* __restrict__ w,
                                                   __bf16* __restrict__ wb, int n) {
    int i = (blockIdx.x * 256 + threadIdx.x) * 4;
    if (i < n) {
        float4 v = *(const float4*)(w + i);
        bf16x4 o;
        o[0] = (__bf16)v.x; o[1] = (__bf16)v.y; o[2] = (__bf16)v.z; o[3] = (__bf16)v.w;
        *(bf16x4*)(wb + i) = o;
    }
}

// ---------------------------------------------------------------------------
// Kernel 2: c2[b][a] = b_enc[a] + b_dec[a] + dot(dec_hidden[b], W_dec[a])
// ---------------------------------------------------------------------------
__global__ __launch_bounds__(256) void k_c2(const float* __restrict__ dh,
                                            const float* __restrict__ Wd,
                                            const float* __restrict__ be,
                                            const float* __restrict__ bd,
                                            float* __restrict__ c2) {
    const int b = blockIdx.x, t = threadIdx.x;
    __shared__ float dsh[ATT_DIMM];
    dsh[t]       = dh[b * ATT_DIMM + t];
    dsh[t + 256] = dh[b * ATT_DIMM + t + 256];
    __syncthreads();
    float a0 = be[t] + bd[t];
    float a1 = be[t + 256] + bd[t + 256];
    const float4* r0 = (const float4*)(Wd + (size_t)t * ATT_DIMM);
    const float4* r1 = (const float4*)(Wd + (size_t)(t + 256) * ATT_DIMM);
    for (int k = 0; k < ATT_DIMM / 4; ++k) {
        float4 w0 = r0[k], w1 = r1[k];
        float d0 = dsh[4 * k], d1 = dsh[4 * k + 1], d2 = dsh[4 * k + 2], d3 = dsh[4 * k + 3];
        a0 = fmaf(w0.x, d0, fmaf(w0.y, d1, fmaf(w0.z, d2, fmaf(w0.w, d3, a0))));
        a1 = fmaf(w1.x, d0, fmaf(w1.y, d1, fmaf(w1.z, d2, fmaf(w1.w, d3, a1))));
    }
    c2[b * ATT_DIMM + t]       = a0;
    c2[b * ATT_DIMM + t + 256] = a1;
}

// ---------------------------------------------------------------------------
// Kernel 3: fused  att[m] = sum_n W_full[n] * relu( (enc @ W_enc^T)[m,n] + c2[b(m),n] )
// 784 blocks x 256 threads (8 waves).
//  - B tile (512 x 128 bf16, padded rows) pulled L2->LDS by the Tensor Data Mover
//  - A tile (64 x 128) staged f32->bf16 by VALU while the TDM streams B
//  - each wave: 4 M-strips x 4 N-tiles register blocking (4x operand reuse)
// ---------------------------------------------------------------------------
__global__ __launch_bounds__(256) void k_gemm_att(const float* __restrict__ enc,
                                                  const __bf16* __restrict__ Wb,
                                                  const float* __restrict__ c2,
                                                  const float* __restrict__ wfull,
                                                  float* __restrict__ att) {
    __shared__ __bf16 As[MTILE][LDS_STR];          // 17,408 B
    __shared__ __bf16 Bs[ATT_DIMM][LDS_STR];       // 139,264 B
    __shared__ float  attpart[8][MTILE];           // 2,048 B   (total 158,720 B)

    const int tid  = threadIdx.x;
    const int lane = tid & 31;
    const int wave = tid >> 5;
    const int m0   = blockIdx.x * MTILE;
    const int half = lane >> 4;   // hi/lo 16 lanes
    const int l16  = lane & 15;
    const int n0   = wave * 64;   // each wave owns 64 columns (4 N-tiles)

    f32x8 acc[4][4];
#pragma unroll
    for (int s = 0; s < 4; ++s)
#pragma unroll
        for (int t = 0; t < 4; ++t)
#pragma unroll
            for (int v = 0; v < 8; ++v) acc[s][t][v] = 0.0f;

    // LDS byte offset of Bs for the TDM descriptor (flat -> addrspace(3) cast)
    const unsigned ldsB =
        (unsigned)(unsigned long long)(__attribute__((address_space(3))) char*)(&Bs[0][0]);
    const unsigned long long wbase = (unsigned long long)(const void*)Wb;

    for (int kc = 0; kc < ENC_DIMM; kc += KC) {
        // ---- TDM: async L2->LDS copy of B tile [512 rows x 128 bf16], issued by wave 0.
        // D# group0: count=1, lds_addr, global_addr (tile start), type=2.
        // D# group1: data_size=2B (code 1), pad_enable, pad_interval=64 dwords (code 5),
        //            pad_amount=4 dwords (code 3)  ->  LDS row stride = 136 bf16,
        //            tensor_dim0=2048, tensor_dim1=512, tile_dim0=128, tile_dim1=512,
        //            tensor_dim0_stride=2048 elements.
        if (wave == 0) {
            unsigned long long ga = wbase + (unsigned long long)kc * 2ull;
            v4u g0;
            g0[0] = 1u;                                                   // count=1
            g0[1] = ldsB;                                                 // lds_addr
            g0[2] = (unsigned)ga;                                         // global_addr lo
            g0[3] = (unsigned)((ga >> 32) & 0x01FFFFFFu) | 0x80000000u;   // addr hi | type=2
            v8i g1;
            g1[0] = (1 << 16) | (1 << 20) | (5 << 22) | (3 << 25);        // dsz|pad en/int/amt
            g1[1] = (int)((2048u & 0xFFFFu) << 16);                       // tensor_dim0 lo
            g1[2] = (int)(512u << 16);                                    // dim0 hi | dim1 lo
            g1[3] = (int)(128u << 16);                                    // dim1 hi | tile_dim0
            g1[4] = 512;                                                  // tile_dim1 | tile_dim2=0
            g1[5] = 2048;                                                 // dim0_stride lo
            g1[6] = 0;
            g1[7] = 0;
            v4i gz4 = {0, 0, 0, 0};
            v8i gz8 = {0, 0, 0, 0, 0, 0, 0, 0};
            __builtin_amdgcn_tensor_load_to_lds(g0, g1, gz4, gz4, gz8, 0);
        }

        // ---- stage A: 64 x 128 f32 -> bf16 into LDS (coalesced; overlaps the TDM) ----
#pragma unroll
        for (int i = 0; i < 8; ++i) {
            int l   = tid + 256 * i;   // 0..2047 float4 slots
            int row = l >> 5;
            int c4  = l & 31;
            float4 v = *(const float4*)(enc + (size_t)(m0 + row) * ENC_DIMM + kc + c4 * 4);
            bf16x4 o;
            o[0] = (__bf16)v.x; o[1] = (__bf16)v.y; o[2] = (__bf16)v.z; o[3] = (__bf16)v.w;
            *(bf16x4*)&As[row][c4 * 4] = o;
        }

        if (wave == 0) __builtin_amdgcn_s_wait_tensorcnt(0);
        __syncthreads();   // A stores + B TDM both visible to all waves

#pragma unroll
        for (int kk = 0; kk < KC; kk += 32) {
            // A fragments (16x32 bf16): lane<16 -> K kk..kk+15; lane>=16 -> K kk+16..kk+31
            bf16x16 a[4];
            const int kbA = kk + half * 16;
#pragma unroll
            for (int s = 0; s < 4; ++s) {
                bf16x8 lo = *(const bf16x8*)&As[s * 16 + l16][kbA];
                bf16x8 hi = *(const bf16x8*)&As[s * 16 + l16][kbA + 8];
#pragma unroll
                for (int i = 0; i < 8; ++i) { a[s][i] = lo[i]; a[s][i + 8] = hi[i]; }
            }
            // B fragments (32x16 bf16): col n = l16, lane<16 -> K 0..15, lane>=16 -> K 16..31
            bf16x16 bfr[4];
            const int kbB = kk + half * 16;
#pragma unroll
            for (int t = 0; t < 4; ++t)
                bfr[t] = *(const bf16x16*)&Bs[n0 + t * 16 + l16][kbB];

#pragma unroll
            for (int s = 0; s < 4; ++s)
#pragma unroll
                for (int t = 0; t < 4; ++t)
                    acc[s][t] = __builtin_amdgcn_wmma_f32_16x16x32_bf16(
                        false, a[s], false, bfr[t], (short)0, acc[s][t], false, false);
        }
        __syncthreads();
    }

    // ---- fused epilogue: +c2[b(m),n], relu, dot with W_full[n], reduce over n ----
    // C/D layout: lane 0-15: N=lane, M=v ; lane 16-31: N=lane-16, M=v+8
#pragma unroll
    for (int s = 0; s < 4; ++s) {
        const int rbase = m0 + s * 16 + half * 8;
        unsigned bv[8];
#pragma unroll
        for (int v = 0; v < 8; ++v) bv[v] = (unsigned)(rbase + v) / 196u;

        float attp[8];
#pragma unroll
        for (int v = 0; v < 8; ++v) attp[v] = 0.0f;

#pragma unroll
        for (int t = 0; t < 4; ++t) {
            const int n   = n0 + t * 16 + l16;
            const float wf = wfull[n];
#pragma unroll
            for (int v = 0; v < 8; ++v) {
                float cv = c2[bv[v] * 512u + n];
                float h  = acc[s][t][v] + cv;
                h = h > 0.0f ? h : 0.0f;
                attp[v] = fmaf(wf, h, attp[v]);
            }
        }
#pragma unroll
        for (int v = 0; v < 8; ++v) {
            float r = attp[v];
            r += __shfl_xor(r, 1, 32);
            r += __shfl_xor(r, 2, 32);
            r += __shfl_xor(r, 4, 32);
            r += __shfl_xor(r, 8, 32);   // sum over the 16-lane N group
            if (l16 == 0) attpart[wave][s * 16 + half * 8 + v] = r;
        }
    }
    __syncthreads();
    if (tid < MTILE) {
        float r = 0.0f;
#pragma unroll
        for (int w = 0; w < 8; ++w) r += attpart[w][tid];
        att[m0 + tid] = r;
    }
}

// ---------------------------------------------------------------------------
// Kernel 4: per-batch softmax over P=196 (b_full dropped: shift-invariant)
//           + alpha-weighted sum over enc (second, coalesced pass over enc)
// ---------------------------------------------------------------------------
__global__ __launch_bounds__(256) void k_soft_wsum(const float* __restrict__ enc,
                                                   const float* __restrict__ att,
                                                   float* __restrict__ out_enc,
                                                   float* __restrict__ out_alpha) {
    const int b = blockIdx.x, t = threadIdx.x;
    __shared__ float red[256];
    __shared__ float al[PIX];

    float v = (t < PIX) ? att[b * PIX + t] : -3.0e38f;
    red[t] = v; __syncthreads();
    for (int s = 128; s > 0; s >>= 1) { if (t < s) red[t] = fmaxf(red[t], red[t + s]); __syncthreads(); }
    const float mx = red[0]; __syncthreads();

    float e = (t < PIX) ? __expf(v - mx) : 0.0f;
    red[t] = e; __syncthreads();
    for (int s = 128; s > 0; s >>= 1) { if (t < s) red[t] += red[t + s]; __syncthreads(); }
    const float inv = 1.0f / red[0];
    if (t < PIX) { float a = e * inv; al[t] = a; out_alpha[b * PIX + t] = a; }
    __syncthreads();

    const float4* eb = (const float4*)(enc + (size_t)b * PIX * ENC_DIMM);
    float4 a0 = {0, 0, 0, 0}, a1 = {0, 0, 0, 0};
    for (int p = 0; p < PIX; ++p) {
        const float ap = al[p];
        float4 x0 = eb[(size_t)p * 512 + t];
        float4 x1 = eb[(size_t)p * 512 + t + 256];
        a0.x = fmaf(ap, x0.x, a0.x); a0.y = fmaf(ap, x0.y, a0.y);
        a0.z = fmaf(ap, x0.z, a0.z); a0.w = fmaf(ap, x0.w, a0.w);
        a1.x = fmaf(ap, x1.x, a1.x); a1.y = fmaf(ap, x1.y, a1.y);
        a1.z = fmaf(ap, x1.z, a1.z); a1.w = fmaf(ap, x1.w, a1.w);
    }
    float4* ob = (float4*)(out_enc + (size_t)b * ENC_DIMM);
    ob[t] = a0; ob[t + 256] = a1;
}

// ---------------------------------------------------------------------------
extern "C" void kernel_launch(void* const* d_in, const int* in_sizes, int n_in,
                              void* d_out, int out_size, void* d_ws, size_t ws_size,
                              hipStream_t stream) {
    const float* enc   = (const float*)d_in[0];   // [256,196,2048]
    const float* dech  = (const float*)d_in[1];   // [256,512]
    const float* Wenc  = (const float*)d_in[2];   // [512,2048]
    const float* benc  = (const float*)d_in[3];   // [512]
    const float* Wdec  = (const float*)d_in[4];   // [512,512]
    const float* bdec  = (const float*)d_in[5];   // [512]
    const float* Wfull = (const float*)d_in[6];   // [1,512]
    // d_in[7] = b_full: unused (softmax is shift-invariant; cancels in both outputs)

    char* ws = (char*)d_ws;
    __bf16* Wb  = (__bf16*)ws;                                            // 2 MB
    float*  c2  = (float*)(ws + (size_t)ATT_DIMM * ENC_DIMM * 2);         // 512 KB
    float*  att = (float*)(ws + (size_t)ATT_DIMM * ENC_DIMM * 2
                               + (size_t)BATCH * ATT_DIMM * 4);           // 200 KB

    float* out_enc   = (float*)d_out;                       // [256,2048]
    float* out_alpha = out_enc + (size_t)BATCH * ENC_DIMM;  // [256,196]

    k_wenc_bf16<<<(ATT_DIMM * ENC_DIMM) / 1024, 256, 0, stream>>>(Wenc, Wb, ATT_DIMM * ENC_DIMM);
    k_c2<<<BATCH, 256, 0, stream>>>(dech, Wdec, benc, bdec, c2);
    k_gemm_att<<<MROWS / MTILE, 256, 0, stream>>>(enc, Wb, c2, Wfull, att);
    k_soft_wsum<<<BATCH, 256, 0, stream>>>(enc, att, out_enc, out_alpha);
}
